// GRURegressor_85469849191004
// MI455X (gfx1250) — matrix operand order repaired and checked by reading, not compiled, not run on hardware
//
#include <hip/hip_runtime.h>
#include <math.h>

// ---------------------------------------------------------------------------
// GRU regressor on MI455X (gfx1250), single persistent workgroup.
//
// Roofline: recurrence = 1500 sequential steps x ~20 MFLOP -> latency bound.
// One WGP, 16 wave32s, all state resident in LDS (283KB of the 320KB budget):
//   W_hh f16 [384][128]  96KB   (B operand, row-major = contiguous K loads)
//   W_ih f16 [384][96]   72KB   (K padded 80->96 with zeros)
//   h    f16 [2][128][128] 64KB (double buffered -> 1 barrier per step)
//   x_t  f16 [2][128][96]  48KB (double buffered, prefetched during GEMM)
// 21 v_wmma_f32_16x16x32_f16 per (16x16 m,c)-cell, 64 cells/step = 1344/step.
// Each wave owns cells covering all 3 gates of the same hidden columns, so
// sigmoid/tanh gate math happens on register-resident f32 accumulators and
// the f32 master copy of h never leaves VGPRs.
// ---------------------------------------------------------------------------

#define BSZ 128
#define TLEN 1500
#define NMEL 80
#define KX 96          // padded input K (multiple of 32)
#define HID 128
#define G3 384         // 3 * HID

typedef __attribute__((ext_vector_type(16))) _Float16 v16h;
typedef __attribute__((ext_vector_type(8)))  _Float16 v8h;
typedef __attribute__((ext_vector_type(4)))  _Float16 v4h;
typedef __attribute__((ext_vector_type(8)))  float    v8f;

__device__ __forceinline__ v8f wmma_f16(v16h a, v16h b, v8f c) {
  // D = A(16x32 f16) x B(32x16 f16) + C(16x16 f32)
  return __builtin_amdgcn_wmma_f32_16x16x32_f16(
      /*neg_a=*/false, a, /*neg_b=*/false, b,
      /*c_mod=*/(short)0, c, /*reuse_a=*/false, /*reuse_b=*/false);
}

// A operand: 16x32 f16 tile from row-major [row][stride] f16 in LDS.
// lane<16: M=lane, K contiguous k0..k0+15
// lane>=16: M=lane-16, K contiguous k0+16..k0+31
__device__ __forceinline__ v16h load_a(const _Float16* __restrict__ base,
                                       int stride, int row0, int k0) {
  const int lane = threadIdx.x & 31;
  const int r = lane & 15;
  const int kk = k0 + ((lane >> 4) << 4);
  const _Float16* p = base + (row0 + r) * stride + kk;
  v8h lo = *(const v8h*)(p);        // ds_load_b128
  v8h hi = *(const v8h*)(p + 8);    // ds_load_b128
  v16h a;
#pragma unroll
  for (int i = 0; i < 8; ++i) { a[i] = lo[i]; a[i + 8] = hi[i]; }
  return a;
}

// B operand: 32x16 f16 tile; B[k][n] = W[n][k] with W row-major [n][stride].
// lane picks column n = n0 + lane%16, contiguous 16 halfs of K starting at
// k0 + 16*(lane/16)  -> two ds_load_b128.
__device__ __forceinline__ v16h load_b(const _Float16* __restrict__ base,
                                       int stride, int n0, int k0) {
  const int lane = threadIdx.x & 31;
  const int n = n0 + (lane & 15);
  const int kk = k0 + ((lane >> 4) << 4);
  const _Float16* p = base + n * stride + kk;
  v8h lo = *(const v8h*)(p);
  v8h hi = *(const v8h*)(p + 8);
  v16h b;
#pragma unroll
  for (int i = 0; i < 8; ++i) { b[i] = lo[i]; b[i + 8] = hi[i]; }
  return b;
}

__device__ __forceinline__ float sigmoidf_(float v) {
  return 1.0f / (1.0f + __expf(-v));
}

__global__ __launch_bounds__(512, 1)
void gru_regressor_persistent(const float* __restrict__ x,
                              const float* __restrict__ wih_f,
                              const float* __restrict__ whh_f,
                              const float* __restrict__ bih_f,
                              const float* __restrict__ bhh_f,
                              const float* __restrict__ wih_b,
                              const float* __restrict__ bih_b,
                              const float* __restrict__ bhh_b,
                              const float* __restrict__ w1,
                              const float* __restrict__ b1,
                              const float* __restrict__ w2,
                              const float* __restrict__ b2,
                              float* __restrict__ out) {
  __shared__ _Float16 s_Whh[G3 * HID];      // 96KB, row-major [n][k]
  __shared__ _Float16 s_Wih[G3 * KX];       // 72KB, row-major [n][k], zero pad
  __shared__ _Float16 s_h[2][BSZ * HID];    // 64KB, double buffered
  __shared__ _Float16 s_x[2][BSZ * KX];     // 48KB, double buffered
  __shared__ float    s_bih[G3];
  __shared__ float    s_bhh[G3];

  const int tid  = threadIdx.x;
  const int wave = tid >> 5;
  const int lane = tid & 31;

  // ------------------------- one-time init -------------------------------
  for (int i = tid; i < G3 * HID; i += 512) s_Whh[i] = (_Float16)whh_f[i];
  for (int i = tid; i < G3 * KX; i += 512) {
    int n = i / KX, k = i - n * KX;
    s_Wih[i] = (k < NMEL) ? (_Float16)wih_f[n * NMEL + k] : (_Float16)0.0f;
  }
  for (int i = tid; i < G3; i += 512) { s_bih[i] = bih_f[i]; s_bhh[i] = bhh_f[i]; }
  for (int i = tid; i < 2 * BSZ * HID; i += 512) ((_Float16*)s_h)[i] = (_Float16)0.0f;
  for (int i = tid; i < 2 * BSZ * KX; i += 512) ((_Float16*)s_x)[i] = (_Float16)0.0f;

  // x_0 -> s_x[0]; 4 threads per batch row, 20 floats each (80 cols total)
  const int brow = tid >> 2;
  const int c0   = (tid & 3) * 20;
  {
    const float* p = x + ((size_t)brow * TLEN + 0) * NMEL + c0;
#pragma unroll
    for (int g = 0; g < 5; ++g) {
      float4 v = *(const float4*)(p + 4 * g);
      v4h hv;
      hv[0] = (_Float16)v.x; hv[1] = (_Float16)v.y;
      hv[2] = (_Float16)v.z; hv[3] = (_Float16)v.w;
      *(v4h*)(&s_x[0][brow * KX + c0 + 4 * g]) = hv;
    }
  }

  // f32 master copy of h: each lane owns the 32 elements it produces.
  float hreg[4][8];
#pragma unroll
  for (int q = 0; q < 4; ++q)
#pragma unroll
    for (int i = 0; i < 8; ++i) hreg[q][i] = 0.0f;

  int cur = 0;

  // ------------------------- recurrence ----------------------------------
  for (int t = 0; t < TLEN; ++t) {
    __syncthreads();  // h_{t-1} stores + x_t stores visible

    // issue global loads of x_{t+1} early; LDS stores deferred past the GEMMs
    float4 xr[5];
    const bool pf = (t + 1) < TLEN;
    if (pf) {
      const float* p = x + ((size_t)brow * TLEN + (t + 1)) * NMEL + c0;
#pragma unroll
      for (int g = 0; g < 5; ++g) xr[g] = *(const float4*)(p + 4 * g);
    }

    const _Float16* hin = s_h[cur];
    const _Float16* xin = s_x[t & 1];
    const int nxt = cur ^ 1;

#pragma unroll
    for (int q = 0; q < 4; ++q) {
      const int cid = (wave << 2) + q;  // 0..63
      const int m = cid >> 3;           // batch tile 0..7
      const int c = cid & 7;            // hidden-col tile 0..7
      v8f accR = {}, accZ = {}, accNx = {}, accNh = {};

#pragma unroll
      for (int kk = 0; kk < KX; kk += 32) {      // x @ W_ih^T (3 K-steps)
        v16h ax = load_a(xin, KX, m * 16, kk);
        accR  = wmma_f16(ax, load_b(s_Wih, KX, 0 * HID + c * 16, kk), accR);
        accZ  = wmma_f16(ax, load_b(s_Wih, KX, 1 * HID + c * 16, kk), accZ);
        accNx = wmma_f16(ax, load_b(s_Wih, KX, 2 * HID + c * 16, kk), accNx);
      }
#pragma unroll
      for (int kk = 0; kk < HID; kk += 32) {     // h @ W_hh^T (4 K-steps)
        v16h ah = load_a(hin, HID, m * 16, kk);
        accR  = wmma_f16(ah, load_b(s_Whh, HID, 0 * HID + c * 16, kk), accR);
        accZ  = wmma_f16(ah, load_b(s_Whh, HID, 1 * HID + c * 16, kk), accZ);
        accNh = wmma_f16(ah, load_b(s_Whh, HID, 2 * HID + c * 16, kk), accNh);
      }

      // gate math (C layout: VGPR i -> row m*16 + i + 8*(lane>=16), col lane%16)
      const int col = c * 16 + (lane & 15);
      const float bir = s_bih[col],           bhr = s_bhh[col];
      const float biz = s_bih[HID + col],     bhz = s_bhh[HID + col];
      const float bin = s_bih[2 * HID + col], bhn = s_bhh[2 * HID + col];
      const int rowb = m * 16 + ((lane >> 4) << 3);
#pragma unroll
      for (int i = 0; i < 8; ++i) {
        float r  = sigmoidf_(accR[i] + bir + bhr);
        float z  = sigmoidf_(accZ[i] + biz + bhz);
        float n  = tanhf(accNx[i] + bin + r * (accNh[i] + bhn));
        float hn = (1.0f - z) * n + z * hreg[q][i];
        hreg[q][i] = hn;
        s_h[nxt][(rowb + i) * HID + col] = (_Float16)hn;
      }
    }

    // commit prefetched x_{t+1}
    if (pf) {
      _Float16* dst = &s_x[(t + 1) & 1][brow * KX + c0];
#pragma unroll
      for (int g = 0; g < 5; ++g) {
        v4h hv;
        hv[0] = (_Float16)xr[g].x; hv[1] = (_Float16)xr[g].y;
        hv[2] = (_Float16)xr[g].z; hv[3] = (_Float16)xr[g].w;
        *(v4h*)(dst + 4 * g) = hv;
      }
    }

    cur ^= 1;
  }
  // forward final hidden h_f lives in s_h[cur]

  // ---------------- backward direction: ONE step from h0=0 ---------------
  // gh = h0 @ W_hh_b^T + b_hh_b = b_hh_b  -> w_hh_b is not needed at all.
  __syncthreads();
  for (int i = tid; i < G3 * KX; i += 512) {
    int n = i / KX, k = i - n * KX;
    s_Wih[i] = (k < NMEL) ? (_Float16)wih_b[n * NMEL + k] : (_Float16)0.0f;
  }
  for (int i = tid; i < G3; i += 512) { s_bih[i] = bih_b[i]; s_bhh[i] = bhh_b[i]; }
  {
    const float* p = x + ((size_t)brow * TLEN + (TLEN - 1)) * NMEL + c0;
#pragma unroll
    for (int g = 0; g < 5; ++g) {
      float4 v = *(const float4*)(p + 4 * g);
      v4h hv;
      hv[0] = (_Float16)v.x; hv[1] = (_Float16)v.y;
      hv[2] = (_Float16)v.z; hv[3] = (_Float16)v.w;
      *(v4h*)(&s_x[0][brow * KX + c0 + 4 * g]) = hv;
    }
  }
  __syncthreads();

  const int hbb = cur ^ 1;  // store h_b in the other h buffer
#pragma unroll
  for (int q = 0; q < 4; ++q) {
    const int cid = (wave << 2) + q;
    const int m = cid >> 3;
    const int c = cid & 7;
    v8f accR = {}, accZ = {}, accNx = {};
#pragma unroll
    for (int kk = 0; kk < KX; kk += 32) {
      v16h ax = load_a(s_x[0], KX, m * 16, kk);
      accR  = wmma_f16(ax, load_b(s_Wih, KX, 0 * HID + c * 16, kk), accR);
      accZ  = wmma_f16(ax, load_b(s_Wih, KX, 1 * HID + c * 16, kk), accZ);
      accNx = wmma_f16(ax, load_b(s_Wih, KX, 2 * HID + c * 16, kk), accNx);
    }
    const int col = c * 16 + (lane & 15);
    const float bir = s_bih[col],           bhr = s_bhh[col];
    const float biz = s_bih[HID + col],     bhz = s_bhh[HID + col];
    const float bin = s_bih[2 * HID + col], bhn = s_bhh[2 * HID + col];
    const int rowb = m * 16 + ((lane >> 4) << 3);
#pragma unroll
    for (int i = 0; i < 8; ++i) {
      float r  = sigmoidf_(accR[i] + bir + bhr);
      float z  = sigmoidf_(accZ[i] + biz + bhz);
      float n  = tanhf(accNx[i] + bin + r * bhn);  // gh_n = b_hh_n (h0 = 0)
      float hb = (1.0f - z) * n;                   // + z*h0 = 0
      s_h[hbb][(rowb + i) * HID + col] = (_Float16)hb;
    }
  }
  __syncthreads();

  // ---------------- head: relu(last @ w1^T + b1) @ w2^T + b2 --------------
  if (tid < BSZ) {
    const _Float16* hf = s_h[cur];
    const _Float16* hb = s_h[hbb];
    float acc = b2[0];
    for (int d = 0; d < 64; ++d) {
      float s = b1[d];
      const float* wr = w1 + d * 256;
      for (int j = 0; j < HID; ++j) s += (float)hf[tid * HID + j] * wr[j];
      for (int j = 0; j < HID; ++j) s += (float)hb[tid * HID + j] * wr[HID + j];
      acc += fmaxf(s, 0.0f) * w2[d];
    }
    out[tid] = acc;
  }
}

extern "C" void kernel_launch(void* const* d_in, const int* in_sizes, int n_in,
                              void* d_out, int out_size, void* d_ws, size_t ws_size,
                              hipStream_t stream) {
  (void)in_sizes; (void)n_in; (void)out_size; (void)d_ws; (void)ws_size;
  const float* x     = (const float*)d_in[0];
  const float* wih_f = (const float*)d_in[1];
  const float* whh_f = (const float*)d_in[2];
  const float* bih_f = (const float*)d_in[3];
  const float* bhh_f = (const float*)d_in[4];
  const float* wih_b = (const float*)d_in[5];
  // d_in[6] (w_hh_b) mathematically unused: backward runs one step from h0=0
  const float* bih_b = (const float*)d_in[7];
  const float* bhh_b = (const float*)d_in[8];
  const float* w1    = (const float*)d_in[9];
  const float* b1    = (const float*)d_in[10];
  const float* w2    = (const float*)d_in[11];
  const float* b2    = (const float*)d_in[12];

  hipLaunchKernelGGL(gru_regressor_persistent, dim3(1), dim3(512), 0, stream,
                     x, wih_f, whh_f, bih_f, bhh_f, wih_b, bih_b, bhh_b,
                     w1, b1, w2, b2, (float*)d_out);
}
